// MultiHeadSelfAttention_987842478919
// MI455X (gfx1250) — compile-verified, matrix operand order checked by reading, not run on hardware
//
#include <hip/hip_runtime.h>
#include <hip/hip_fp16.h>
#include <stdint.h>

typedef __attribute__((ext_vector_type(16))) _Float16 v16h;
typedef __attribute__((ext_vector_type(8)))  _Float16 v8h;
typedef __attribute__((ext_vector_type(8)))  float    v8f;
typedef __attribute__((ext_vector_type(4)))  float    v4f;
typedef __attribute__((ext_vector_type(4)))  int      v4i;

#define D_MODEL 1024
#define NHEADS  16
#define DKH     64
#define BATCH   2
#define SEQ     2048
#define MTOT    (BATCH * SEQ)   // 4096

// ---- gfx1250 async global->LDS copy (ASYNCcnt path), with safe fallback ----
#if defined(__has_builtin)
#if __has_builtin(__builtin_amdgcn_global_load_async_to_lds_b128)
#define HAVE_ASYNC_LDS 1
#endif
#endif

#if defined(HAVE_ASYNC_LDS)
// Signature (from clang diagnostic): (v4i AS1* gsrc, v4i AS3* ldst, imm, imm)
#define ASYNC_CP16(ldst, gsrc)                                                 \
  __builtin_amdgcn_global_load_async_to_lds_b128(                              \
      (__attribute__((address_space(1))) v4i*)(uintptr_t)(gsrc),               \
      (__attribute__((address_space(3))) v4i*)(uintptr_t)(ldst), 0, 0)
#define ASYNC_WAIT() asm volatile("s_wait_asynccnt 0" ::: "memory")
#else
#define ASYNC_CP16(ldst, gsrc) (*(v8h*)(ldst) = *(const v8h*)(gsrc))
#define ASYNC_WAIT()
#endif

// ---------------- WMMA fragment helpers (ISA 7.12.2 layouts) ----------------
__device__ __forceinline__ v16h pack16(v8h lo, v8h hi) {
  v16h r;
#pragma unroll
  for (int e = 0; e < 8; ++e) { r[e] = lo[e]; r[e + 8] = hi[e]; }
  return r;
}

// rowbase points at the 32-wide K-chunk start of this lane's row.
// Lane (half, m): holds K = {half*8..half*8+7} and {16+half*8..16+half*8+7}.
__device__ __forceinline__ v16h frag_ld(const _Float16* rowbase, int hg) {
  v8h lo = *(const v8h*)(rowbase + hg * 8);
  v8h hi = *(const v8h*)(rowbase + 16 + hg * 8);
  return pack16(lo, hi);
}

__device__ __forceinline__ v8f wmma32(v16h a, v16h b, v8f c) {
  return __builtin_amdgcn_wmma_f32_16x16x32_f16(false, a, false, b, (short)0, c,
                                                false, false);
}

// ---------------- GEMM: C[M,N] = A[M,1024] * W[1024,N] + bias ----------------
#define GBM 128
#define GBN 128
#define GBK 64
#define LDT 72   // LDS row stride in halves (64 + 8 pad, keeps 16B alignment)

template <bool A_HALF, bool QKV_OUT>
__device__ __forceinline__ void gemm_body(const void* __restrict__ Ap,
                                          const float* __restrict__ W,
                                          const float* __restrict__ bias,
                                          void* __restrict__ Dst,
                                          _Float16* As, _Float16* Bs) {
  const int tid  = threadIdx.x;
  const int wave = tid >> 5, lane = tid & 31;
  const int hg   = lane >> 4, lm = lane & 15;
  const int bm   = blockIdx.y * GBM, bn = blockIdx.x * GBN;
  const int wm   = (wave >> 1) * 32;  // 4 waves along M
  const int wn   = (wave & 1) * 64;   // 2 waves along N

  v8f acc[2][4];
#pragma unroll
  for (int i = 0; i < 2; ++i)
#pragma unroll
    for (int j = 0; j < 4; ++j)
#pragma unroll
      for (int e = 0; e < 8; ++e) acc[i][j][e] = 0.f;

  for (int k0 = 0; k0 < D_MODEL; k0 += GBK) {
    __syncthreads();
    // ---- A tile -> LDS, row-major [128][64] f16 ----
    if (A_HALF) {
      const _Float16* A = (const _Float16*)Ap;
#pragma unroll
      for (int i = 0; i < 4; ++i) {
        int c = tid + i * 256;              // 1024 b128 chunks
        int row = c >> 3, col = (c & 7) * 8;
        ASYNC_CP16(As + row * LDT + col, A + (bm + row) * D_MODEL + k0 + col);
      }
    } else {
      const float* A = (const float*)Ap;
#pragma unroll
      for (int i = 0; i < 8; ++i) {
        int s = tid + i * 256;              // 2048 float4 slots
        int row = s >> 4, c4 = (s & 15) * 4;
        v4f v = *(const v4f*)(A + (bm + row) * D_MODEL + k0 + c4);
#pragma unroll
        for (int j = 0; j < 4; ++j) As[row * LDT + c4 + j] = (_Float16)v[j];
      }
    }
    // ---- W tile -> LDS transposed: Bs[n][k] ----
#pragma unroll
    for (int i = 0; i < 8; ++i) {
      int s = tid + i * 256;                // 2048 float4 slots (64k x 128n)
      int k = s >> 5, n4 = (s & 31) * 4;
      v4f v = *(const v4f*)(W + (k0 + k) * D_MODEL + bn + n4);
#pragma unroll
      for (int j = 0; j < 4; ++j) Bs[(n4 + j) * LDT + k] = (_Float16)v[j];
    }
    if (k0 + GBK < D_MODEL)  // gfx1250 global_prefetch_b8 for next W tile
      __builtin_prefetch((const void*)(W + (k0 + GBK) * D_MODEL + bn), 0, 0);
    ASYNC_WAIT();
    __syncthreads();

#pragma unroll
    for (int kk = 0; kk < 2; ++kk) {
      v16h a0 = frag_ld(As + (wm + 0 * 16 + lm) * LDT + kk * 32, hg);
      v16h a1 = frag_ld(As + (wm + 1 * 16 + lm) * LDT + kk * 32, hg);
      v16h bf[4];
#pragma unroll
      for (int j = 0; j < 4; ++j)
        bf[j] = frag_ld(Bs + (wn + j * 16 + lm) * LDT + kk * 32, hg);
#pragma unroll
      for (int j = 0; j < 4; ++j) {
        acc[0][j] = wmma32(a0, bf[j], acc[0][j]);
        acc[1][j] = wmma32(a1, bf[j], acc[1][j]);
      }
    }
  }

  // ---- epilogue: C row = half*8 + r, col = lane&15 (ISA C/D layout) ----
#pragma unroll
  for (int i = 0; i < 2; ++i) {
#pragma unroll
    for (int j = 0; j < 4; ++j) {
      int mg = bm + wm + i * 16 + hg * 8;
      int ng = bn + wn + j * 16 + lm;
      float bval = bias[ng];
      if (QKV_OUT) {
        _Float16* D = (_Float16*)Dst;
        int h = ng >> 6, dk = ng & 63;
#pragma unroll
        for (int r = 0; r < 8; ++r) {
          int m = mg + r;
          int bidx = m >> 11, s = m & (SEQ - 1);
          D[((bidx * NHEADS + h) * SEQ + s) * DKH + dk] =
              (_Float16)(acc[i][j][r] + bval);
        }
      } else {
        float* D = (float*)Dst;
#pragma unroll
        for (int r = 0; r < 8; ++r)
          D[(mg + r) * D_MODEL + ng] = acc[i][j][r] + bval;
      }
    }
  }
}

__global__ void __launch_bounds__(256) qkv_gemm_kernel(
    const float* x, const float* wq, const float* bq, _Float16* dq,
    const float* wk, const float* bk, _Float16* dk_, const float* wv,
    const float* bv, _Float16* dv) {
  __shared__ _Float16 As[GBM * LDT];
  __shared__ _Float16 Bs[GBN * LDT];
  const float* W;
  const float* b;
  _Float16* d;
  if (blockIdx.z == 0)      { W = wq; b = bq; d = dq; }
  else if (blockIdx.z == 1) { W = wk; b = bk; d = dk_; }
  else                      { W = wv; b = bv; d = dv; }
  gemm_body<false, true>(x, W, b, d, As, Bs);
}

__global__ void __launch_bounds__(256) out_gemm_kernel(const _Float16* att,
                                                       const float* wo,
                                                       const float* bo,
                                                       float* out) {
  __shared__ _Float16 As[GBM * LDT];
  __shared__ _Float16 Bs[GBN * LDT];
  gemm_body<true, false>(att, wo, bo, out, As, Bs);
}

// ---------------- Flash attention: one (b,h) x 128-query tile per block -----
#define LDK 72   // 64 + 8 pad halves

__global__ void __launch_bounds__(256) attn_kernel(const _Float16* __restrict__ Q,
                                                   const _Float16* __restrict__ K,
                                                   const _Float16* __restrict__ V,
                                                   _Float16* __restrict__ Aout) {
  __shared__ _Float16 Ks[64 * LDK];          // [key][dk]   (B-frag for QK^T)
  __shared__ _Float16 Vt[64 * LDK];          // [dk][key]   (B-frag for PV)
  __shared__ _Float16 Ps[8 * 16 * LDK];      // per-wave P scratch [16][64]

  const int tid  = threadIdx.x;
  const int wave = tid >> 5, lane = tid & 31;
  const int hg   = lane >> 4, lm = lane & 15;
  const int bh   = blockIdx.y;               // b*16 + h
  const int q0   = blockIdx.x * 128;

  // Persistent Q fragments: this wave owns query rows q0+wave*16 .. +15.
  const _Float16* Qb = Q + (bh * SEQ + q0 + wave * 16) * DKH;
  v16h aq0 = frag_ld(Qb + lm * DKH + 0,  hg);
  v16h aq1 = frag_ld(Qb + lm * DKH + 32, hg);

  float m_run[8], l_run[8];
  v8f o[4];
#pragma unroll
  for (int r = 0; r < 8; ++r) { m_run[r] = -3.0e38f; l_run[r] = 0.f; }
#pragma unroll
  for (int j = 0; j < 4; ++j)
#pragma unroll
    for (int e = 0; e < 8; ++e) o[j][e] = 0.f;

  const _Float16* Kb = K + bh * SEQ * DKH;
  const _Float16* Vb = V + bh * SEQ * DKH;
  _Float16* Pw = Ps + wave * 16 * LDK;

  for (int kt = 0; kt < SEQ / 64; ++kt) {
    const int key0 = kt * 64;
    __syncthreads();
    {  // K tile: straight copy (already [n=key][k=dk]) via async-to-LDS
      int row = tid >> 2, cb = (tid & 3) * 16;
      const _Float16* g = Kb + (key0 + row) * DKH + cb;
      ASYNC_CP16(Ks + row * LDK + cb,     g);
      ASYNC_CP16(Ks + row * LDK + cb + 8, g + 8);
    }
    {  // V tile transposed: Vt[dk][key]
#pragma unroll
      for (int i = 0; i < 2; ++i) {
        int c = tid + i * 256;               // 512 v8h chunks
        int key = c >> 3, d0 = (c & 7) * 8;
        v8h v = *(const v8h*)(Vb + (key0 + key) * DKH + d0);
#pragma unroll
        for (int j = 0; j < 8; ++j) Vt[(d0 + j) * LDK + key] = v[j];
      }
    }
    ASYNC_WAIT();
    __syncthreads();

    // S = Q * K^T  (contract over dk=64 -> two K=32 WMMA steps)
    v8f sf[4];
#pragma unroll
    for (int j = 0; j < 4; ++j) {
      v8f c;
#pragma unroll
      for (int e = 0; e < 8; ++e) c[e] = 0.f;
      v16h b0 = frag_ld(Ks + (j * 16 + lm) * LDK + 0,  hg);
      v16h b1 = frag_ld(Ks + (j * 16 + lm) * LDK + 32, hg);
      c = wmma32(aq0, b0, c);
      c = wmma32(aq1, b1, c);
      sf[j] = c;
    }

    // Online softmax: row (hg*8+r) spans 16 lanes of this half-group.
    float fscale[8];
#pragma unroll
    for (int r = 0; r < 8; ++r) {
      float tm = -3.0e38f;
#pragma unroll
      for (int j = 0; j < 4; ++j) {
        sf[j][r] *= 0.125f;                  // 1/sqrt(64)
        tm = fmaxf(tm, sf[j][r]);
      }
#pragma unroll
      for (int off = 1; off < 16; off <<= 1)
        tm = fmaxf(tm, __shfl_xor(tm, off, 32));
      float nm = fmaxf(m_run[r], tm);
      fscale[r] = __expf(m_run[r] - nm);
      m_run[r]  = nm;
      float rs = 0.f;
#pragma unroll
      for (int j = 0; j < 4; ++j) {
        float p = __expf(sf[j][r] - nm);
        sf[j][r] = p;
        rs += p;
      }
#pragma unroll
      for (int off = 1; off < 16; off <<= 1) rs += __shfl_xor(rs, off, 32);
      l_run[r] = l_run[r] * fscale[r] + rs;
    }

    // Rescale accumulated O by exp(m_old - m_new)
#pragma unroll
    for (int j = 0; j < 4; ++j)
#pragma unroll
      for (int r = 0; r < 8; ++r) o[j][r] *= fscale[r];

    // C-layout -> A-layout reshape of P through per-wave LDS.
#pragma unroll
    for (int j = 0; j < 4; ++j)
#pragma unroll
      for (int r = 0; r < 8; ++r)
        Pw[(hg * 8 + r) * LDK + j * 16 + lm] = (_Float16)sf[j][r];
    // Wave-local LDS fence (CDNA5 split counter): writes above, reads below.
    asm volatile("s_wait_dscnt 0" ::: "memory");

    v16h ap0 = frag_ld(Pw + lm * LDK + 0,  hg);
    v16h ap1 = frag_ld(Pw + lm * LDK + 32, hg);
#pragma unroll
    for (int j = 0; j < 4; ++j) {
      v16h b0 = frag_ld(Vt + (j * 16 + lm) * LDK + 0,  hg);
      v16h b1 = frag_ld(Vt + (j * 16 + lm) * LDK + 32, hg);
      o[j] = wmma32(ap0, b0, o[j]);
      o[j] = wmma32(ap1, b1, o[j]);
    }
  }

  // Normalize and store to [B, S, H*Dk] f16 for the output projection.
  const int b = bh >> 4, h = bh & 15;
#pragma unroll
  for (int r = 0; r < 8; ++r) {
    float inv = 1.0f / l_run[r];
    int s = q0 + wave * 16 + hg * 8 + r;
#pragma unroll
    for (int j = 0; j < 4; ++j) {
      int dk = j * 16 + lm;
      Aout[(b * SEQ + s) * D_MODEL + h * DKH + dk] = (_Float16)(o[j][r] * inv);
    }
  }
}

// ---------------------------------------------------------------------------
extern "C" void kernel_launch(void* const* d_in, const int* in_sizes, int n_in,
                              void* d_out, int out_size, void* d_ws,
                              size_t ws_size, hipStream_t stream) {
  const float* x  = (const float*)d_in[0];
  const float* wq = (const float*)d_in[1];
  const float* bq = (const float*)d_in[2];
  const float* wk = (const float*)d_in[3];
  const float* bk = (const float*)d_in[4];
  const float* wv = (const float*)d_in[5];
  const float* bv = (const float*)d_in[6];
  const float* wo = (const float*)d_in[7];
  const float* bo = (const float*)d_in[8];
  float* out = (float*)d_out;

  const size_t per = (size_t)MTOT * D_MODEL;  // elements per f16 tensor
  _Float16* qb = (_Float16*)d_ws;             // [B,H,S,Dk]
  _Float16* kb = qb + per;
  _Float16* vb = kb + per;
  _Float16* ab = vb + per;                    // [B,S,H*Dk]  (total 32 MB)

  dim3 blk(256, 1, 1);
  qkv_gemm_kernel<<<dim3(D_MODEL / GBN, MTOT / GBM, 3), blk, 0, stream>>>(
      x, wq, bq, qb, wk, bk, kb, wv, bv, vb);
  attn_kernel<<<dim3(SEQ / 128, BATCH * NHEADS), blk, 0, stream>>>(qb, kb, vb,
                                                                   ab);
  out_gemm_kernel<<<dim3(D_MODEL / GBN, MTOT / GBM), blk, 0, stream>>>(ab, wo,
                                                                       bo, out);
}
